// GRAND_48112223650339
// MI455X (gfx1250) — matrix operand order repaired and checked by reading, not compiled, not run on hardware
//
#include <hip/hip_runtime.h>

#define NN    8192
#define HDIM  64
#define BATCH 2
#define FIN   128

typedef __attribute__((ext_vector_type(16))) __bf16 v16bf;
typedef __attribute__((ext_vector_type(8)))  __bf16 v8bf;
typedef __attribute__((ext_vector_type(8)))  float  v8f;

static __device__ inline v16bf cat8(v8bf lo, v8bf hi) {
  return __builtin_shufflevector(lo, hi, 0,1,2,3,4,5,6,7,8,9,10,11,12,13,14,15);
}

// A-fragment (16x32 bf16, M x K) per-lane loader from row-major bf16.
// lanes 0-15: v0..7 = K[kk..kk+15]
// lanes16-31: v0..7 = K[kk+16..kk+31]
static __device__ inline v16bf loadA_bf16(const __bf16* row, int kk, int half) {
  v8bf lo = *(const v8bf*)(row + kk + half * 16);
  v8bf hi = *(const v8bf*)(row + kk + 8 + half * 16);
  return cat8(lo, hi);
}

static __device__ inline v16bf loadA_f32(const float* row, int kk, int half) {
  v8f f0 = *(const v8f*)(row + kk + half * 16);
  v8f f1 = *(const v8f*)(row + kk + 8 + half * 16);
  v16bf r;
#pragma unroll
  for (int i = 0; i < 8; ++i) { r[i] = (__bf16)f0[i]; r[8 + i] = (__bf16)f1[i]; }
  return r;
}

// B-fragment (32x16 bf16, K x N): lane holds 16 consecutive K of one column.
static __device__ inline v16bf loadB_bf16(const __bf16* p) {
  v8bf lo = *(const v8bf*)p;
  v8bf hi = *(const v8bf*)(p + 8);
  return cat8(lo, hi);
}

static __device__ inline v16bf loadB_f32(const float* p) {
  v8f f0 = *(const v8f*)p;
  v8f f1 = *(const v8f*)(p + 8);
  v16bf r;
#pragma unroll
  for (int i = 0; i < 8; ++i) { r[i] = (__bf16)f0[i]; r[8 + i] = (__bf16)f1[i]; }
  return r;
}

// -------------------- adj f32 -> bf16 (one batch slice) ----------------------
__global__ void GRAND_cvt_bf16(const float* __restrict__ src,
                               __bf16* __restrict__ dst, long long n) {
  long long i = ((long long)blockIdx.x * blockDim.x + threadIdx.x) * 8;
  if (i + 8 <= n) {
    v8f f = *(const v8f*)(src + i);
    v8bf o;
#pragma unroll
    for (int r = 0; r < 8; ++r) o[r] = (__bf16)f[r];
    *(v8bf*)(dst + i) = o;
  }
}

// -------------------- diffusion: out = adj @ h  (N x N) x (N x H) ------------
// hT / outT stored transposed [HDIM][NN] so B-fragments are contiguous.
// outF ([NN][HDIM], f32) written only when writeF != 0 (last hop of a layer).
// Launched with grid.y == 1; all pointers pre-offset to the current batch.
template <bool ABF16>
__global__ __launch_bounds__(128)
void GRAND_diffuse(const void* __restrict__ adjP,
                   const __bf16* __restrict__ hT,
                   __bf16* __restrict__ outT,
                   float* __restrict__ outF,
                   int writeF) {
  const int lane = threadIdx.x & 31;
  const int wave = threadIdx.x >> 5;
  const int half = lane >> 4;
  const int ln   = lane & 15;
  const int m0   = blockIdx.x * 128 + wave * 32;   // 32 rows per wave

  const __bf16* aB = (const __bf16*)adjP;
  const float*  aF = (const float*)adjP;

  const size_t r0 = (size_t)(m0 + ln) * NN;
  const size_t r1 = (size_t)(m0 + 16 + ln) * NN;

  const v8f vzero = {0.f, 0.f, 0.f, 0.f, 0.f, 0.f, 0.f, 0.f};
  v8f acc[2][4];
#pragma unroll
  for (int i = 0; i < 2; ++i)
#pragma unroll
    for (int j = 0; j < 4; ++j) acc[i][j] = vzero;

  for (int kk = 0; kk < NN; kk += 32) {
    v16bf a0, a1;
    if (ABF16) {
      a0 = loadA_bf16(aB + r0, kk, half);
      a1 = loadA_bf16(aB + r1, kk, half);
      __builtin_prefetch(aB + r0 + kk + 1024, 0, 0);  // global_prefetch_b8
      __builtin_prefetch(aB + r1 + kk + 1024, 0, 0);
    } else {
      a0 = loadA_f32(aF + r0, kk, half);
      a1 = loadA_f32(aF + r1, kk, half);
    }
    const int kb = kk + half * 16;
#pragma unroll
    for (int nt = 0; nt < 4; ++nt) {
      v16bf bf = loadB_bf16(hT + (size_t)(nt * 16 + ln) * NN + kb);
      acc[0][nt] = __builtin_amdgcn_wmma_f32_16x16x32_bf16(
          false, a0, false, bf, (short)0, acc[0][nt], false, false);
      acc[1][nt] = __builtin_amdgcn_wmma_f32_16x16x32_bf16(
          false, a1, false, bf, (short)0, acc[1][nt], false, false);
    }
  }

  // C/D layout: lane (ln, half) owns column ln, rows half*8 + r of each tile
#pragma unroll
  for (int i = 0; i < 2; ++i) {
    const int mi = m0 + i * 16;
#pragma unroll
    for (int nt = 0; nt < 4; ++nt) {
      const int g = nt * 16 + ln;
      v8bf o;
#pragma unroll
      for (int r = 0; r < 8; ++r) o[r] = (__bf16)acc[i][nt][r];
      *(v8bf*)(outT + (size_t)g * NN + mi + half * 8) = o;
      if (writeF) {
#pragma unroll
        for (int r = 0; r < 8; ++r)
          outF[(size_t)(mi + half * 8 + r) * HDIM + g] = acc[i][nt][r];
      }
    }
  }
}

// ------------- linear: out = A @ W^T + bias (+A residual) (+relu) ------------
// A: [NN][KD] f32 row-major, W: [HDIM][KD] f32 row-major. grid.y == 1.
template <int KD>
__global__ __launch_bounds__(128)
void GRAND_linear(const float* __restrict__ A,
                  const float* __restrict__ W,
                  const float* __restrict__ bias,
                  float* __restrict__ outF,
                  __bf16* __restrict__ outT,
                  int addRes, int doRelu, int writeF, int writeT) {
  const int lane = threadIdx.x & 31;
  const int wave = threadIdx.x >> 5;
  const int half = lane >> 4;
  const int ln   = lane & 15;
  const int m0   = blockIdx.x * 128 + wave * 32;

  const size_t r0 = (size_t)(m0 + ln) * KD;
  const size_t r1 = (size_t)(m0 + 16 + ln) * KD;

  const v8f vzero = {0.f, 0.f, 0.f, 0.f, 0.f, 0.f, 0.f, 0.f};
  v8f acc[2][4];
#pragma unroll
  for (int i = 0; i < 2; ++i)
#pragma unroll
    for (int j = 0; j < 4; ++j) acc[i][j] = vzero;

#pragma unroll
  for (int kk = 0; kk < KD; kk += 32) {
    v16bf a0 = loadA_f32(A + r0, kk, half);
    v16bf a1 = loadA_f32(A + r1, kk, half);
    const int kb = kk + half * 16;
#pragma unroll
    for (int nt = 0; nt < 4; ++nt) {
      // B[k][n=g] = W[g][k] -> contiguous in k for fixed column g
      v16bf bf = loadB_f32(W + (size_t)(nt * 16 + ln) * KD + kb);
      acc[0][nt] = __builtin_amdgcn_wmma_f32_16x16x32_bf16(
          false, a0, false, bf, (short)0, acc[0][nt], false, false);
      acc[1][nt] = __builtin_amdgcn_wmma_f32_16x16x32_bf16(
          false, a1, false, bf, (short)0, acc[1][nt], false, false);
    }
  }

#pragma unroll
  for (int i = 0; i < 2; ++i) {
    const int mi = m0 + i * 16;
#pragma unroll
    for (int nt = 0; nt < 4; ++nt) {
      const int g  = nt * 16 + ln;
      const float bg = bias[g];
      float v[8];
#pragma unroll
      for (int r = 0; r < 8; ++r) {
        float t = acc[i][nt][r] + bg;
        if (addRes && KD == HDIM)
          t += A[(size_t)(mi + half * 8 + r) * KD + g];
        if (doRelu) t = fmaxf(t, 0.f);
        v[r] = t;
      }
      if (writeT) {
        v8bf o;
#pragma unroll
        for (int r = 0; r < 8; ++r) o[r] = (__bf16)v[r];
        *(v8bf*)(outT + (size_t)g * NN + mi + half * 8) = o;
      }
      if (writeF) {
#pragma unroll
        for (int r = 0; r < 8; ++r)
          outF[(size_t)(mi + half * 8 + r) * HDIM + g] = v[r];
      }
    }
  }
}

extern "C" void kernel_launch(void* const* d_in, const int* in_sizes, int n_in,
                              void* d_out, int out_size, void* d_ws, size_t ws_size,
                              hipStream_t stream) {
  (void)in_sizes; (void)n_in; (void)out_size;
  const float* x   = (const float*)d_in[0];
  const float* adj = (const float*)d_in[1];
  const float* W0  = (const float*)d_in[2];
  const float* b0  = (const float*)d_in[3];
  const float* Ws  = (const float*)d_in[4];
  const float* bs  = (const float*)d_in[5];
  float* out = (float*)d_out;

  // Workspace: single-batch buffers; batches processed sequentially so the
  // 128 MB bf16 adjacency slice stays resident in the 192 MB L2 across all
  // 8 hops of that batch (only hop 1 streams from HBM).
  char* ws = (char*)d_ws;
  const size_t hT_bytes = (size_t)HDIM * NN * sizeof(__bf16);   // 1 MB
  const size_t dF_bytes = (size_t)NN * HDIM * sizeof(float);    // 2 MB
  __bf16* hTa = (__bf16*)(ws);
  __bf16* hTb = (__bf16*)(ws + hT_bytes);
  float*  dF  = (float*)(ws + 2 * hT_bytes);
  const size_t adjWsOff = 2 * hT_bytes + dF_bytes;              // 4 MB
  const size_t adjSlice = (size_t)NN * NN;                      // per-batch elems
  __bf16* adjb = (__bf16*)(ws + adjWsOff);
  const bool useAdjB = ws_size >= adjWsOff + adjSlice * sizeof(__bf16);

  const dim3 blk(128);
  const dim3 grd(NN / 128, 1);

  for (int b = 0; b < BATCH; ++b) {
    const float* adjF = adj + (size_t)b * adjSlice;

    if (useAdjB) {
      const long long n = (long long)adjSlice;
      const int threads = 256;
      const unsigned blocks = (unsigned)((n / 8 + threads - 1) / threads);
      GRAND_cvt_bf16<<<dim3(blocks), dim3(threads), 0, stream>>>(adjF, adjb, n);
    }

    // h0 = x @ W0^T + b0 -> transposed bf16 (feeds layer-0 diffusion)
    GRAND_linear<FIN><<<grd, blk, 0, stream>>>(
        x + (size_t)b * NN * FIN, W0, b0,
        out + (size_t)b * NN * HDIM, hTa,
        /*res*/0, /*relu*/0, /*wF*/0, /*wT*/1);

    __bf16* cur = hTa;
    __bf16* nxt = hTb;
    for (int layer = 0; layer < 2; ++layer) {
      for (int s = 0; s < 4; ++s) {
        const int last = (s == 3) ? 1 : 0;  // last hop also writes f32 diffusion
        if (useAdjB)
          GRAND_diffuse<true><<<grd, blk, 0, stream>>>(adjb, cur, nxt, dF, last);
        else
          GRAND_diffuse<false><<<grd, blk, 0, stream>>>(adjF, cur, nxt, dF, last);
        __bf16* t = cur; cur = nxt; nxt = t;
      }
      if (layer == 0) {
        // h = relu(dF @ Ws0^T + bs0) -> transposed bf16 for layer-1 diffusion
        GRAND_linear<HDIM><<<grd, blk, 0, stream>>>(
            dF, Ws, bs, out + (size_t)b * NN * HDIM, cur,
            /*res*/0, /*relu*/1, /*wF*/0, /*wT*/1);
      } else {
        // h = dF @ Ws1^T + bs1 + dF -> final f32 output for this batch
        GRAND_linear<HDIM><<<grd, blk, 0, stream>>>(
            dF, Ws + HDIM * HDIM, bs + HDIM, out + (size_t)b * NN * HDIM, cur,
            /*res*/1, /*relu*/0, /*wF*/1, /*wT*/0);
      }
    }
  }
}
